// _Transcriber_67302137528857
// MI455X (gfx1250) — matrix operand order repaired and checked by reading, not compiled, not run on hardware
//
#include <hip/hip_runtime.h>
#include <hip/hip_bf16.h>

typedef __attribute__((ext_vector_type(16))) _Float16 v16h;
typedef __attribute__((ext_vector_type(8)))  _Float16 v8h;
typedef __attribute__((ext_vector_type(8)))  float    v8f;

#define BB   8
#define TIN  2052
#define DIN  80
#define DD   512
#define HH   8
#define HD   64
#define SEQ  640
#define RR   128
#define UU   512
#define FFND 2048
#define LL   20
#define MROWS (BB*SEQ)      // 5120
#define NEGV (-1.0e9f)

// ---------------------------------------------------------------------------
// fragment load helpers (CDNA5 WMMA f16 16x16x32 layouts, wave32)
// A: row-major, lane m=lane&15, hi=lane>>4 ; k = kk + hi*16 + e (16 contiguous halves)
// B: stored as N x K ("Bt"), lane n=lane&15 ; k = kk + hi*16 + e (16 contiguous halves)
// ---------------------------------------------------------------------------
__device__ inline v16h load_fragA(const _Float16* rowp, int kk, int hi) {
    v8h lo = *(const v8h*)(rowp + kk + hi * 16);
    v8h hh = *(const v8h*)(rowp + kk + hi * 16 + 8);
    v16h r;
#pragma unroll
    for (int i = 0; i < 8; ++i) { r[i] = lo[i]; r[8 + i] = hh[i]; }
    return r;
}
__device__ inline v16h load_fragB(const _Float16* colp) {
    v8h lo = *(const v8h*)(colp);
    v8h hh = *(const v8h*)(colp + 8);
    v16h r;
#pragma unroll
    for (int i = 0; i < 8; ++i) { r[i] = lo[i]; r[8 + i] = hh[i]; }
    return r;
}

// ---------------------------------------------------------------------------
// fp32 -> f16 transposed weight conversion: W (K x N) -> O (N x K)
// ---------------------------------------------------------------------------
__global__ void k_convT(const float* __restrict__ W, _Float16* __restrict__ O,
                        int K, int N) {
    int idx = blockIdx.x * blockDim.x + threadIdx.x;
    if (idx >= K * N) return;
    int n = idx / K, k = idx % K;
    O[idx] = (_Float16)W[(size_t)k * N + n];
}

// ---------------------------------------------------------------------------
// fused input projection + reshape + rc/utt gather -> cat (B, 640, 512) fp32
// ---------------------------------------------------------------------------
__global__ void k_inproj(const float* __restrict__ in, const float* __restrict__ w,
                         float* __restrict__ cat) {
    int idx = blockIdx.x * blockDim.x + threadIdx.x;
    if (idx >= BB * SEQ * DD) return;
    int c  = idx % DD;
    int p  = (idx / DD) % SEQ;
    int b  = idx / (DD * SEQ);
    int t  = (p < RR) ? ((p < RR - 1) ? (p + 1) * 4 : 512) : (p - RR);
    int row = t * 4 + (c >> 7);          // source row in (B, 2052, 80)
    int cc  = c & 127;                   // col in (80 -> 128) projection
    const float* ip = in + ((size_t)b * TIN + row) * DIN;
    float s = 0.f;
#pragma unroll
    for (int d = 0; d < DIN; ++d) s += ip[d] * w[d * 128 + cc];
    cat[idx] = s;
}

// ---------------------------------------------------------------------------
// wave-per-row LayerNorm; writes fp32 and/or f16
// ---------------------------------------------------------------------------
__global__ void k_layernorm(const float* __restrict__ X, const float* __restrict__ g,
                            const float* __restrict__ b, float* __restrict__ Yf,
                            _Float16* __restrict__ Yh, int rows, int D) {
    int lane = threadIdx.x & 31;
    int row  = blockIdx.x * (blockDim.x >> 5) + (threadIdx.x >> 5);
    if (row >= rows) return;
    const float* x = X + (size_t)row * D;
    float s = 0.f, s2 = 0.f;
    for (int i = lane; i < D; i += 32) { float v = x[i]; s += v; s2 += v * v; }
#pragma unroll
    for (int off = 16; off; off >>= 1) {
        s  += __shfl_xor(s,  off, 32);
        s2 += __shfl_xor(s2, off, 32);
    }
    float mean = s / D;
    float var  = s2 / D - mean * mean;
    float inv  = rsqrtf(var + 1e-5f);
    for (int i = lane; i < D; i += 32) {
        float v = (x[i] - mean) * inv * g[i] + b[i];
        if (Yf) Yf[(size_t)row * D + i] = v;
        if (Yh) Yh[(size_t)row * D + i] = (_Float16)v;
    }
}

// ---------------------------------------------------------------------------
// generic WMMA GEMM: C = A(MxK,f16,row) * Bt(NxK,f16)^T + bias [+res] [relu]
// block = 128 threads (4 waves); each wave computes GM x GN 16x16 tiles
// (2x4 = 32 rows x 64 cols); block covers 32 rows x 256 cols.
// grid = (N/256, M/32). __launch_bounds__(128,1) -> full VGPR budget, no spills.
// ---------------------------------------------------------------------------
#define GM 2
#define GN 4
__global__ void __launch_bounds__(128, 1)
k_gemm(const _Float16* __restrict__ A, const _Float16* __restrict__ Bt,
       const float* __restrict__ bias, const float* __restrict__ res,
       float* __restrict__ Cf, _Float16* __restrict__ Ch,
       int M, int N, int K, int relu) {
    int lane = threadIdx.x & 31;
    int wave = threadIdx.x >> 5;
    int hi   = lane >> 4;
    int ln15 = lane & 15;
    int tile_m0 = blockIdx.y * GM;
    int tile_n0 = (blockIdx.x * 4 + wave) * GN;

    const _Float16* Arow[GM];
#pragma unroll
    for (int i = 0; i < GM; ++i)
        Arow[i] = A + (size_t)((tile_m0 + i) * 16 + ln15) * K;
    const _Float16* Brow[GN];
#pragma unroll
    for (int j = 0; j < GN; ++j)
        Brow[j] = Bt + (size_t)((tile_n0 + j) * 16 + ln15) * K + hi * 16;

    v8f acc[GM][GN];
#pragma unroll
    for (int i = 0; i < GM; ++i)
#pragma unroll
        for (int j = 0; j < GN; ++j)
            acc[i][j] = (v8f){0.f,0.f,0.f,0.f,0.f,0.f,0.f,0.f};

    for (int kk = 0; kk < K; kk += 32) {
        v16h af[GM];
        v16h bf[GN];
#pragma unroll
        for (int i = 0; i < GM; ++i) af[i] = load_fragA(Arow[i], kk, hi);
#pragma unroll
        for (int j = 0; j < GN; ++j) bf[j] = load_fragB(Brow[j] + kk);
#pragma unroll
        for (int i = 0; i < GM; ++i)
#pragma unroll
            for (int j = 0; j < GN; ++j)
                acc[i][j] = __builtin_amdgcn_wmma_f32_16x16x32_f16(
                    false, af[i], false, bf[j], (short)0, acc[i][j], false, false);
    }

#pragma unroll
    for (int i = 0; i < GM; ++i) {
#pragma unroll
        for (int j = 0; j < GN; ++j) {
            int n = (tile_n0 + j) * 16 + ln15;
            float bv = bias ? bias[n] : 0.f;
#pragma unroll
            for (int r = 0; r < 8; ++r) {
                int mm = (tile_m0 + i) * 16 + hi * 8 + r;
                float v = acc[i][j][r] + bv;
                if (res)  v += res[(size_t)mm * N + n];
                if (relu) v = fmaxf(v, 0.f);
                if (Cf) Cf[(size_t)mm * N + n] = v;
                if (Ch) Ch[(size_t)mm * N + n] = (_Float16)v;
            }
        }
    }
}

// ---------------------------------------------------------------------------
// V transpose: kv (B*640, 1024) f16, V = cols [512,1024) -> vt (B, H, 64, 640)
// ---------------------------------------------------------------------------
__global__ void k_transpose_v(const _Float16* __restrict__ kv, _Float16* __restrict__ vt) {
    int idx = blockIdx.x * blockDim.x + threadIdx.x;
    if (idx >= BB * HH * HD * SEQ) return;
    int pos = idx % SEQ;
    int d   = (idx / SEQ) % HD;
    int h   = (idx / (SEQ * HD)) % HH;
    int b   = idx / (SEQ * HD * HH);
    vt[idx] = kv[((size_t)b * SEQ + pos) * (2 * DD) + DD + h * HD + d];
}

// ---------------------------------------------------------------------------
// Emformer mask (SEG=4, RCL=1, LC=30, R=128, U=512) evaluated analytically
// ---------------------------------------------------------------------------
__device__ inline float emf_mask(float v, int i, int j, int lenr) {
    int s = (i < RR) ? i : ((i - RR) >> 2);
    bool ok = (j == s);
    if (!ok) {
        int lo = s * 4 - 30; if (lo < 0) lo = 0;
        int hi2 = (s + 1) * 4; if (hi2 > UU) hi2 = UU;
        ok = (j >= RR + lo) && (j < RR + hi2);
    }
    if (j >= RR + lenr) ok = false;
    return ok ? v : NEGV;
}

// ---------------------------------------------------------------------------
// fused attention: per (qtile, head, batch): S = QK^T*scale -> mask -> softmax
// -> P*V. block = 128 (4 waves). LDS scores fp32 + probs f16.
// ---------------------------------------------------------------------------
#define SW 656   // 640 + pad
__global__ void __launch_bounds__(128, 1)
k_attention(const _Float16* __restrict__ q,
            const _Float16* __restrict__ kv,
            const _Float16* __restrict__ vt,
            const int* __restrict__ lengths,
            _Float16* __restrict__ att) {
    __shared__ float    sS[16][SW];
    __shared__ _Float16 sP[16][SW];

    int lane = threadIdx.x & 31;
    int wave = threadIdx.x >> 5;
    int hi   = lane >> 4;
    int ln15 = lane & 15;
    int qt   = blockIdx.x;           // 0..39
    int h    = blockIdx.y;           // 0..7
    int bb   = blockIdx.z;           // 0..7
    int lenr = lengths[bb] >> 2;     // lengths // STRIDE
    const float scale = 0.125f;      // 64^-0.5

    // ---- scores: Q (16x64) x K^T -> 16 x 640 ----
    const _Float16* Qrow = q + ((size_t)bb * SEQ + qt * 16 + ln15) * DD + h * HD;
    v16h a0 = load_fragA(Qrow, 0, hi);
    v16h a1 = load_fragA(Qrow, 32, hi);

    for (int nt = wave; nt < SEQ / 16; nt += 4) {
        int n = nt * 16 + ln15;
        const _Float16* Krow = kv + ((size_t)bb * SEQ + n) * (2 * DD) + h * HD;
        v8f acc = (v8f){0.f,0.f,0.f,0.f,0.f,0.f,0.f,0.f};
        v16h b0 = load_fragB(Krow + 0 + hi * 16);
        v16h b1 = load_fragB(Krow + 32 + hi * 16);
        acc = __builtin_amdgcn_wmma_f32_16x16x32_f16(false, a0, false, b0, (short)0, acc, false, false);
        acc = __builtin_amdgcn_wmma_f32_16x16x32_f16(false, a1, false, b1, (short)0, acc, false, false);
#pragma unroll
        for (int r = 0; r < 8; ++r) {
            int li = hi * 8 + r;                 // local row
            int gi = qt * 16 + li;               // global query row
            sS[li][n] = emf_mask(acc[r] * scale, gi, n, lenr);
        }
    }
    __syncthreads();

    // ---- softmax over 640 per row; wave w owns rows 4w..4w+3 ----
    for (int lr = wave * 4; lr < wave * 4 + 4; ++lr) {
        float mx = NEGV;
        for (int j = lane; j < SEQ; j += 32) mx = fmaxf(mx, sS[lr][j]);
#pragma unroll
        for (int off = 16; off; off >>= 1) mx = fmaxf(mx, __shfl_xor(mx, off, 32));
        float sum = 0.f;
        for (int j = lane; j < SEQ; j += 32) {
            float e = __expf(sS[lr][j] - mx);
            sum += e;
            sS[lr][j] = e;
        }
#pragma unroll
        for (int off = 16; off; off >>= 1) sum += __shfl_xor(sum, off, 32);
        float rinv = 1.f / sum;
        for (int j = lane; j < SEQ; j += 32)
            sP[lr][j] = (_Float16)(sS[lr][j] * rinv);
    }
    __syncthreads();

    // ---- P (16x640) x V (640x64): wave w -> output dim tile w*16 ----
    int d0 = wave * 16;
    v8f acc = (v8f){0.f,0.f,0.f,0.f,0.f,0.f,0.f,0.f};
    const _Float16* vtb = vt + ((size_t)(bb * HH + h) * HD + d0 + ln15) * SEQ;
    for (int kk = 0; kk < SEQ; kk += 32) {
        v16h a = load_fragA(&sP[ln15][0], kk, hi);
        v16h bfr = load_fragB(vtb + kk + hi * 16);
        acc = __builtin_amdgcn_wmma_f32_16x16x32_f16(false, a, false, bfr, (short)0, acc, false, false);
    }
#pragma unroll
    for (int r = 0; r < 8; ++r) {
        int mm = qt * 16 + hi * 8 + r;
        att[((size_t)bb * SEQ + mm) * DD + h * HD + d0 + ln15] = (_Float16)acc[r];
    }
}

// ---------------------------------------------------------------------------
// pack utt rows (cat[:, 128:, :]) into contiguous f16 (B*512, 512)
// ---------------------------------------------------------------------------
__global__ void k_pack_utt(const float* __restrict__ cat, _Float16* __restrict__ o) {
    int idx = blockIdx.x * blockDim.x + threadIdx.x;
    if (idx >= BB * UU * DD) return;
    int c = idx % DD;
    int r = (idx / DD) % UU;
    int b = idx / (DD * UU);
    o[idx] = (_Float16)cat[((size_t)b * SEQ + RR + r) * DD + c];
}

// lengths_r (int32) bit-cast into tail of the float output buffer
__global__ void k_lengths(const int* __restrict__ lengths, float* __restrict__ tail) {
    int i = threadIdx.x;
    if (i < BB) ((int*)tail)[i] = lengths[i] >> 2;
}

// ---------------------------------------------------------------------------
extern "C" void kernel_launch(void* const* d_in, const int* in_sizes, int n_in,
                              void* d_out, int out_size, void* d_ws, size_t ws_size,
                              hipStream_t stream) {
    const float* input   = (const float*)d_in[0];
    const int*   lengths = (const int*)  d_in[1];
    const float* w_in    = (const float*)d_in[2];
    const float* ln_in_g = (const float*)d_in[3];
    const float* ln_in_b = (const float*)d_in[4];
    const float* wq      = (const float*)d_in[5];
    const float* bq      = (const float*)d_in[6];
    const float* wkv     = (const float*)d_in[7];
    const float* bkv     = (const float*)d_in[8];
    const float* wo      = (const float*)d_in[9];
    const float* bo      = (const float*)d_in[10];
    const float* pff_g   = (const float*)d_in[11];
    const float* pff_b   = (const float*)d_in[12];
    const float* w1      = (const float*)d_in[13];
    const float* b1      = (const float*)d_in[14];
    const float* w2      = (const float*)d_in[15];
    const float* b2      = (const float*)d_in[16];
    const float* lno_g   = (const float*)d_in[17];
    const float* lno_b   = (const float*)d_in[18];
    const float* w_out   = (const float*)d_in[19];
    const float* b_out   = (const float*)d_in[20];
    const float* fln_g   = (const float*)d_in[21];
    const float* fln_b   = (const float*)d_in[22];
    float* out = (float*)d_out;

    // ---- workspace layout ----
    char* p = (char*)d_ws;
    auto alloc = [&](size_t bytes) -> char* {
        char* r = p; p += (bytes + 255) & ~(size_t)255; return r;
    };
    _Float16* wqT  = (_Float16*)alloc((size_t)LL * DD * DD * 2);
    _Float16* wkvT = (_Float16*)alloc((size_t)LL * 2 * DD * DD * 2);
    _Float16* woT  = (_Float16*)alloc((size_t)LL * DD * DD * 2);
    _Float16* w1T  = (_Float16*)alloc((size_t)LL * FFND * DD * 2);
    _Float16* w2T  = (_Float16*)alloc((size_t)LL * DD * FFND * 2);
    _Float16* woutT= (_Float16*)alloc((size_t)1024 * DD * 2);
    float*    cat  = (float*)   alloc((size_t)MROWS * DD * 4);
    _Float16* ln_h = (_Float16*)alloc((size_t)MROWS * DD * 2);
    _Float16* q_h  = (_Float16*)alloc((size_t)MROWS * DD * 2);
    _Float16* kv_h = (_Float16*)alloc((size_t)MROWS * 2 * DD * 2);
    _Float16* vt_h = (_Float16*)alloc((size_t)BB * HH * HD * SEQ * 2);
    _Float16* att_h= (_Float16*)alloc((size_t)MROWS * DD * 2);
    float*    res  = (float*)   alloc((size_t)MROWS * DD * 4);
    _Float16* ff_h = (_Float16*)alloc((size_t)MROWS * DD * 2);
    _Float16* h1_h = (_Float16*)alloc((size_t)MROWS * FFND * 2);
    float*    res2 = (float*)   alloc((size_t)MROWS * DD * 4);
    _Float16* utt_h= (_Float16*)alloc((size_t)BB * UU * DD * 2);
    float*    fin  = (float*)   alloc((size_t)BB * UU * 1024 * 4);

    auto convT = [&](const float* W, _Float16* O, int K, int N) {
        int tot = K * N;
        k_convT<<<dim3((tot + 255) / 256), 256, 0, stream>>>(W, O, K, N);
    };
    // ---- one-time weight conversion (f32 KxN -> f16 NxK) ----
    for (int l = 0; l < LL; ++l) {
        convT(wq + (size_t)l * DD * DD,      wqT  + (size_t)l * DD * DD,      DD, DD);
        convT(wkv + (size_t)l * DD * 2 * DD, wkvT + (size_t)l * 2 * DD * DD,  DD, 2 * DD);
        convT(wo + (size_t)l * DD * DD,      woT  + (size_t)l * DD * DD,      DD, DD);
        convT(w1 + (size_t)l * DD * FFND,    w1T  + (size_t)l * FFND * DD,    DD, FFND);
        convT(w2 + (size_t)l * FFND * DD,    w2T  + (size_t)l * DD * FFND,    FFND, DD);
    }
    convT(w_out, woutT, DD, 1024);

    auto gemm = [&](const _Float16* A, const _Float16* Bt, const float* bias,
                    const float* rs, float* Cf, _Float16* Ch,
                    int M, int N, int K, int relu) {
        k_gemm<<<dim3(N / 256, M / 32), 128, 0, stream>>>(A, Bt, bias, rs, Cf, Ch, M, N, K, relu);
    };
    auto lnorm = [&](const float* X, const float* g, const float* b,
                     float* Yf, _Float16* Yh, int rows, int D) {
        k_layernorm<<<dim3((rows + 7) / 8), 256, 0, stream>>>(X, g, b, Yf, Yh, rows, D);
    };

    // ---- input projection + gather -> cat ----
    {
        int tot = BB * SEQ * DD;
        k_inproj<<<dim3((tot + 255) / 256), 256, 0, stream>>>(input, w_in, cat);
    }

    // ---- 20 encoder layers (cat_{l+1} = layer_out_l) ----
    for (int l = 0; l < LL; ++l) {
        const float* g0 = ln_in_g + (size_t)l * DD;
        const float* c0 = ln_in_b + (size_t)l * DD;
        lnorm(cat, g0, c0, nullptr, ln_h, MROWS, DD);

        gemm(ln_h, wqT  + (size_t)l * DD * DD,     bq  + (size_t)l * DD,     nullptr, nullptr, q_h,  MROWS, DD,     DD,   0);
        gemm(ln_h, wkvT + (size_t)l * 2 * DD * DD, bkv + (size_t)l * 2 * DD, nullptr, nullptr, kv_h, MROWS, 2 * DD, DD,   0);

        {
            int tot = BB * HH * HD * SEQ;
            k_transpose_v<<<dim3((tot + 255) / 256), 256, 0, stream>>>(kv_h, vt_h);
        }
        k_attention<<<dim3(SEQ / 16, HH, BB), 128, 0, stream>>>(q_h, kv_h, vt_h, lengths, att_h);

        gemm(att_h, woT + (size_t)l * DD * DD, bo + (size_t)l * DD, cat, res, nullptr, MROWS, DD, DD, 0);

        lnorm(res, pff_g + (size_t)l * DD, pff_b + (size_t)l * DD, nullptr, ff_h, MROWS, DD);
        gemm(ff_h, w1T + (size_t)l * FFND * DD, b1 + (size_t)l * FFND, nullptr, nullptr, h1_h, MROWS, FFND, DD,   1);
        gemm(h1_h, w2T + (size_t)l * DD * FFND, b2 + (size_t)l * DD,   res,     res2,    nullptr, MROWS, DD, FFND, 0);

        lnorm(res2, lno_g + (size_t)l * DD, lno_b + (size_t)l * DD, cat, nullptr, MROWS, DD);
    }

    // ---- final projection + layernorm ----
    {
        int tot = BB * UU * DD;
        k_pack_utt<<<dim3((tot + 255) / 256), 256, 0, stream>>>(cat, utt_h);
    }
    gemm(utt_h, woutT, b_out, nullptr, fin, nullptr, BB * UU, 1024, DD, 0);
    lnorm(fin, fln_g, fln_b, out, nullptr, BB * UU, 1024);
    k_lengths<<<1, 32, 0, stream>>>(lengths, out + (size_t)BB * UU * 1024);
}
